// DeformableConv2d_36601711296603
// MI455X (gfx1250) — matrix operand order repaired and checked by reading, not compiled, not run on hardware
//
#include <hip/hip_runtime.h>

// ---------------------------------------------------------------------------
// Deformable Conv2d for MI455X (gfx1250, wave32, WMMA, async-to-LDS).
// Pipeline:
//   pack_x      : x (B,C,H,W) f32 -> xt (B,66,66,C) f16, zero-padded border
//   pack_w      : conv_w -> w2 f16, layout [kb][o][32]   (kb = k/32, k = n*256+c)
//   pack_pw     : p_w    -> pw2 f16, layout [kb][j][32], rows 18..31 zero
//   offset_gemm : offs[pix][18] = (pw2 x im2col(xt)) + p_b   (WMMA, M=32)
//   sample      : bilinear sampling -> xoff2 f16, layout [kb][pix][32]
//   main_gemm   : out = w2 x xoff2   (WMMA, M=256,N=16384,K=2304),
//                 double-buffered LDS tiles filled by global_load_async_to_lds_b128,
//                 drained with s_wait_asynccnt (ASYNCcnt path).
// ---------------------------------------------------------------------------

typedef _Float16 v16h __attribute__((ext_vector_type(16)));
typedef _Float16 v8h  __attribute__((ext_vector_type(8)));
typedef float    v8f  __attribute__((ext_vector_type(8)));

constexpr int BATCH = 4;
constexpr int CIN   = 256;
constexpr int HH    = 64;
constexpr int WW    = 64;
constexpr int HP    = 66;   // padded
constexpr int WP    = 66;
constexpr int NTAP  = 9;
constexpr int COUT  = 256;
constexpr int KDIM  = CIN * NTAP;          // 2304
constexpr int NKB   = KDIM / 32;           // 72 k-steps of 32
constexpr int NPIX  = BATCH * HH * WW;     // 16384
constexpr int HWP   = HP * WP;

static __device__ __forceinline__ v16h cat8(v8h lo, v8h hi) {
  return __builtin_shufflevector(lo, hi, 0,1,2,3,4,5,6,7,8,9,10,11,12,13,14,15);
}

// Async copy of 16 bytes/lane from global into LDS (ASYNCcnt-tracked).
// Generic LDS pointer's low 32 bits are the LDS byte address (aperture rule).
static __device__ __forceinline__ void async_copy16(const _Float16* g, _Float16* l) {
  unsigned lds_addr = (unsigned)(size_t)(void*)l;
  unsigned long long gaddr = (unsigned long long)(size_t)g;
  asm volatile("global_load_async_to_lds_b128 %0, %1, off"
               :: "v"(lds_addr), "v"(gaddr)
               : "memory");
}

static __device__ __forceinline__ void wait_async0() {
  asm volatile("s_wait_asynccnt 0x0" ::: "memory");
}

// --------------------------- pack kernels ----------------------------------

__global__ void pack_x_kernel(const float* __restrict__ x, _Float16* __restrict__ xt) {
  int gid = blockIdx.x * 256 + threadIdx.x;           // ((b*66+y)*66+x)*256 + c
  int c  = gid & 255;
  int r  = gid >> 8;
  int xp = r % WP;
  int yp = (r / WP) % HP;
  int b  = r / (WP * HP);
  float v = 0.f;
  if (yp >= 1 && yp <= HH && xp >= 1 && xp <= WW)
    v = x[((b * CIN + c) * HH + (yp - 1)) * WW + (xp - 1)];
  xt[gid] = (_Float16)v;
}

__global__ void pack_w_kernel(const float* __restrict__ cw, _Float16* __restrict__ w2) {
  int gid = blockIdx.x * 256 + threadIdx.x;           // o*KDIM + k
  int k = gid % KDIM;
  int o = gid / KDIM;
  int c = k & 255, n = k >> 8;
  float v = cw[((o * CIN + c) * 3 + n / 3) * 3 + (n % 3)];
  w2[(k >> 5) * (COUT * 32) + o * 32 + (k & 31)] = (_Float16)v;
}

__global__ void pack_pw_kernel(const float* __restrict__ pw, _Float16* __restrict__ pw2) {
  int gid = blockIdx.x * 256 + threadIdx.x;           // j*KDIM + k, j in 0..31
  int k = gid % KDIM;
  int j = gid / KDIM;
  float v = 0.f;
  if (j < 18) {
    int c = k & 255, n = k >> 8;
    v = pw[((j * CIN + c) * 3 + n / 3) * 3 + (n % 3)];
  }
  pw2[(k >> 5) * (32 * 32) + j * 32 + (k & 31)] = (_Float16)v;
}

// --------------------- offset conv: WMMA, M=32 -----------------------------
// A = pw2 [kb][j][32]; B built on the fly from xt taps (im2col fused).
// WG: 8 waves, each wave owns 32 pixel-columns; WG tile 32 x 256.

__global__ void __launch_bounds__(256)
offset_gemm_kernel(const _Float16* __restrict__ pw2, const _Float16* __restrict__ xt,
                   const float* __restrict__ pb, float* __restrict__ offs) {
  int tid  = threadIdx.x;
  int lane = tid & 31, wv = tid >> 5;
  int lcol = lane & 15, hi = lane >> 4;
  int col0 = blockIdx.x * 256 + wv * 32;

  v8f acc[2][2] = {};
  int pix[2], basep[2];
#pragma unroll
  for (int g = 0; g < 2; ++g) {
    int p = col0 + g * 16 + lcol;
    pix[g] = p;
    int b = p >> 12, p2 = p & 4095;
    basep[g] = ((b * HP + (p2 >> 6)) * WP + (p2 & 63)) * CIN;  // tap (0,0) base
  }
  int koff = hi ? 16 : 0;

  for (int kb = 0; kb < NKB; ++kb) {
    const _Float16* abase = pw2 + kb * 1024;
    // A fragments (16x32 f16 layout: per-lane 16 contiguous halves at koff)
    const _Float16* pa0 = abase + lcol * 32 + koff;
    v16h a0 = cat8(*(const v8h*)pa0, *(const v8h*)(pa0 + 8));
    const _Float16* pa1 = abase + (16 + lcol) * 32 + koff;
    v16h a1 = cat8(*(const v8h*)pa1, *(const v8h*)(pa1 + 8));
    // B fragments straight from xt: k -> (tap, channel), 16 contiguous channels
    int k    = kb * 32 + (hi ? 16 : 0);
    int ntap = k >> 8;
    int crem = k & 255;
    int dy = ntap / 3, dx = ntap - dy * 3;
    int tapoff = (dy * WP + dx) * CIN + crem;
    v16h b0 = *(const v16h*)(xt + basep[0] + tapoff);
    v16h b1 = *(const v16h*)(xt + basep[1] + tapoff);

    acc[0][0] = __builtin_amdgcn_wmma_f32_16x16x32_f16(false, a0, false, b0, (short)0, acc[0][0], false, false);
    acc[0][1] = __builtin_amdgcn_wmma_f32_16x16x32_f16(false, a0, false, b1, (short)0, acc[0][1], false, false);
    acc[1][0] = __builtin_amdgcn_wmma_f32_16x16x32_f16(false, a1, false, b0, (short)0, acc[1][0], false, false);
    acc[1][1] = __builtin_amdgcn_wmma_f32_16x16x32_f16(false, a1, false, b1, (short)0, acc[1][1], false, false);
  }

#pragma unroll
  for (int g = 0; g < 2; ++g)
#pragma unroll
    for (int f = 0; f < 2; ++f) {
      int jb = f * 16 + hi * 8;
#pragma unroll
      for (int v = 0; v < 8; ++v) {
        int j = jb + v;
        if (j < 18) offs[pix[g] * 18 + j] = acc[f][g][v] + pb[j];
      }
    }
}

// -------------------------- bilinear sampler -------------------------------
// One block per (pixel, tap); 256 lanes = channels -> fully coalesced gathers.

__global__ void __launch_bounds__(256)
sample_kernel(const float* __restrict__ offs, const _Float16* __restrict__ xt,
              _Float16* __restrict__ xoff2) {
  int pix = blockIdx.x;
  int n   = blockIdx.y;
  int c   = threadIdx.x;
  int b = pix >> 12, p2 = pix & 4095;
  int h = p2 >> 6, w = p2 & 63;

  float offy = offs[pix * 18 + n];
  float offx = offs[pix * 18 + 9 + n];
  float py = (float)(h + n / 3) + offy;     // padded coords: h+1 + (dy-1)
  float px = (float)(w + n % 3) + offx;
  float fy = floorf(py), fx = floorf(px);
  float qlty = fminf(fmaxf(fy, 0.f), 65.f);
  float qltx = fminf(fmaxf(fx, 0.f), 65.f);
  float qrby = fminf(fmaxf(fy + 1.f, 0.f), 65.f);
  float qrbx = fminf(fmaxf(fx + 1.f, 0.f), 65.f);
  py = fminf(fmaxf(py, 0.f), 65.f);
  px = fminf(fmaxf(px, 0.f), 65.f);
  float g_lt = (1.f + (qlty - py)) * (1.f + (qltx - px));
  float g_rb = (1.f - (qrby - py)) * (1.f - (qrbx - px));
  float g_lb = (1.f + (qlty - py)) * (1.f - (qrbx - px));
  float g_rt = (1.f - (qrby - py)) * (1.f + (qltx - px));

  int iy0 = (int)qlty, ix0 = (int)qltx, iy1 = (int)qrby, ix1 = (int)qrbx;
  int base = b * (HWP * CIN) + c;
  float lt = (float)xt[base + (iy0 * WP + ix0) * CIN];
  float rb = (float)xt[base + (iy1 * WP + ix1) * CIN];
  float lb = (float)xt[base + (iy0 * WP + ix1) * CIN];
  float rt = (float)xt[base + (iy1 * WP + ix0) * CIN];
  float v = g_lt * lt + g_rb * rb + g_lb * lb + g_rt * rt;

  int kb = n * 8 + (c >> 5);                // k = n*256 + c
  xoff2[kb * (NPIX * 32) + pix * 32 + (c & 31)] = (_Float16)v;
}

// ----------------------- main GEMM: 256 x 16384 x 2304 ---------------------
// WG tile 256(M) x 64(N); 8 waves in a 4(M) x 2(N) grid, wave tile 64x32.
// Double-buffered LDS tiles, filled with global_load_async_to_lds_b128 and
// drained with s_wait_asynccnt; one barrier per k-step; loads for tile k+1
// overlap the WMMA work of tile k.

__global__ void __launch_bounds__(256)
main_gemm_kernel(const _Float16* __restrict__ w2, const _Float16* __restrict__ xoff2,
                 float* __restrict__ out) {
  __shared__ _Float16 ldsA[2][COUT * 32];   // 2 x 16 KB, layout [o][32]
  __shared__ _Float16 ldsB[2][64 * 32];     // 2 x  4 KB, layout [col][32]

  int tid  = threadIdx.x;
  int lane = tid & 31, wv = tid >> 5;
  int mi = wv & 3, ni = wv >> 2;
  int n0 = blockIdx.x * 64;
  int lcol = lane & 15, hi = lane >> 4;
  int koff = hi ? 16 : 0;

  // per-thread async slots: A tile = 4 chunks of 8 halves, B tile = 1 chunk
  int aoff0 = tid * 8;
  int boff  = tid * 8;

  auto issue_tile = [&](int kb, int buf) {
    const _Float16* ga = w2 + (size_t)kb * (COUT * 32);
    const _Float16* gb = xoff2 + (size_t)kb * (NPIX * 32) + n0 * 32;
#pragma unroll
    for (int i = 0; i < 4; ++i) {
      int off = i * 2048 + aoff0;
      async_copy16(ga + off, &ldsA[buf][off]);
    }
    async_copy16(gb + boff, &ldsB[buf][boff]);
  };

  v8f acc[4][2] = {};
  issue_tile(0, 0);

  for (int kb = 0; kb < NKB; ++kb) {
    int cur = kb & 1;
    wait_async0();       // this wave's writes of buffer `cur` have landed
    __syncthreads();     // all waves' writes visible; prior reads of `cur^1` done
    if (kb + 1 < NKB) issue_tile(kb + 1, cur ^ 1);

    v16h bfr[2];
#pragma unroll
    for (int g = 0; g < 2; ++g) {
      int col = ni * 32 + g * 16 + lcol;
      bfr[g] = *(const v16h*)(&ldsB[cur][0] + col * 32 + hi * 16);
    }
#pragma unroll
    for (int f = 0; f < 4; ++f) {
      int row = mi * 64 + f * 16 + lcol;
      const _Float16* pa = &ldsA[cur][0] + row * 32 + koff;
      v16h a = cat8(*(const v8h*)pa, *(const v8h*)(pa + 8));
#pragma unroll
      for (int g = 0; g < 2; ++g)
        acc[f][g] = __builtin_amdgcn_wmma_f32_16x16x32_f16(
            false, a, false, bfr[g], (short)0, acc[f][g], false, false);
    }
  }

  // epilogue: out (B, Cout, H, W); D rows = out channel, cols = pixels
#pragma unroll
  for (int g = 0; g < 2; ++g) {
    int n = n0 + ni * 32 + g * 16 + lcol;
    int b = n >> 12, p2 = n & 4095;
    float* obase = out + b * (COUT * 4096) + p2;
#pragma unroll
    for (int f = 0; f < 4; ++f) {
      int o0 = mi * 64 + f * 16 + hi * 8;
#pragma unroll
      for (int v = 0; v < 8; ++v)
        obase[(o0 + v) * 4096] = acc[f][g][v];
    }
  }
}

// ------------------------------- launcher ----------------------------------

extern "C" void kernel_launch(void* const* d_in, const int* in_sizes, int n_in,
                              void* d_out, int out_size, void* d_ws, size_t ws_size,
                              hipStream_t stream) {
  const float* x   = (const float*)d_in[0];
  const float* p_w = (const float*)d_in[1];
  const float* p_b = (const float*)d_in[2];
  const float* cw  = (const float*)d_in[3];
  float* out = (float*)d_out;

  char* ws = (char*)d_ws;
  size_t off = 0;
  _Float16* xt = (_Float16*)(ws + off);  off += (size_t)BATCH * HWP * CIN * 2;      // 8,921,088
  _Float16* w2 = (_Float16*)(ws + off);  off += (size_t)COUT * KDIM * 2;            // 1,179,648
  _Float16* pw2 = (_Float16*)(ws + off); off += (size_t)32 * KDIM * 2;              //   147,456
  float*    offs = (float*)(ws + off);   off += (size_t)NPIX * 18 * 4;              // 1,179,648
  _Float16* xoff2 = (_Float16*)(ws + off);                                          // 75,497,472

  pack_x_kernel<<<(BATCH * HWP * CIN) / 256, 256, 0, stream>>>(x, xt);
  pack_w_kernel<<<(COUT * KDIM) / 256, 256, 0, stream>>>(cw, w2);
  pack_pw_kernel<<<(32 * KDIM) / 256, 256, 0, stream>>>(p_w, pw2);
  offset_gemm_kernel<<<NPIX / 256, 256, 0, stream>>>(pw2, xt, p_b, offs);
  dim3 sg(NPIX, NTAP);
  sample_kernel<<<sg, 256, 0, stream>>>(offs, xt, xoff2);
  main_gemm_kernel<<<NPIX / 64, 256, 0, stream>>>(w2, xoff2, out);
}
